// model1_72988674228294
// MI455X (gfx1250) — compile-verified, matrix operand order checked by reading, not run on hardware
//
#include <hip/hip_runtime.h>
#include <hip/hip_bf16.h>
#include <math.h>
#include <stdint.h>

// ============================================================================
// Point-Transformer forward for MI455X (gfx1250, wave32, WMMA).
// B=8, N=512, K=16, NP=128.  All dense GEMMs via v_wmma_f32_16x16x32_f16
// (f16 inputs, f32 accumulate).  Two-layer MLPs are fused through LDS so the
// P x H hidden activations never reach HBM (saves ~256MB of traffic on a1).
// kNN candidate staging uses GLOBAL_LOAD_ASYNC_TO_LDS_B32 (ASYNCcnt path).
// ============================================================================

typedef __attribute__((ext_vector_type(16))) _Float16 v16h;
typedef __attribute__((ext_vector_type(8)))  float    v8f;

#define GEMM_WAVES 4   // 128-thread blocks for single-layer GEMM
#define MLP_WAVES  2   // 64-thread blocks for fused 2-layer MLP (LDS budget)

// ---- WMMA fragment loader --------------------------------------------------
// Loads a 16x32 f16 tile (row-major in LDS, row stride `ld`, ld even) into the
// ISA 7.12.2 A-matrix layout: lane L<16 -> row L, K 0..7 in v0-3 & 16..23 in
// v4-7; lanes 16-31 -> K 8..15 / 24..31.  Used for both A tiles and W^T tiles.
__device__ __forceinline__ v16h load_frag(const _Float16* base, int ld, int lane) {
  union { v16h v; unsigned u[8]; } f;
  const int r  = lane & 15;
  const int kb = (lane >> 4) << 3;             // 0 or 8
  const _Float16* rowp = base + (size_t)r * ld;
#pragma unroll
  for (int vi = 0; vi < 8; ++vi) {
    const int col = (vi < 4) ? (kb + (vi << 1)) : (16 + kb + ((vi - 4) << 1));
    f.u[vi] = *(const unsigned*)(rowp + col);  // ds_load_b32 (aligned: col even)
  }
  return f.v;
}

__device__ __forceinline__ v8f wmma16(v16h a, v16h b, v8f c) {
  return __builtin_amdgcn_wmma_f32_16x16x32_f16(false, a, false, b,
                                                (short)0, c, false, false);
}

__device__ __forceinline__ void lds_fence() {
  asm volatile("s_wait_dscnt 0" ::: "memory");
}

// Async DMA of one b32 element: LDS[ldsAddr] = MEM[saddr + voff] (ASYNCcnt).
__device__ __forceinline__ void async_b32_to_lds(unsigned ldsAddr, unsigned voff,
                                                 const float* saddr) {
  asm volatile("global_load_async_to_lds_b32 %0, %1, %2"
               :: "v"(ldsAddr), "v"(voff), "s"(saddr) : "memory");
}

__device__ __forceinline__ void async_fence() {
  asm volatile("s_wait_asynccnt 0" ::: "memory");
}

// ---- single-layer GEMM: Y[M,N] = act(X[M,K] @ W[K,N] + b) ------------------
__global__ __launch_bounds__(128) void k_gemm1(
    const float* __restrict__ X, const float* __restrict__ W,
    const float* __restrict__ bias, float* __restrict__ Y,
    int M, int K, int N, int Kp, int Np, int relu)
{
  extern __shared__ char smem[];
  float*    s_bias = (float*)smem;                         // Np
  _Float16* s_wt   = (_Float16*)(s_bias + Np);             // Np*Kp (W^T, padded)
  _Float16* s_x    = s_wt + (size_t)Np * Kp;               // GEMM_WAVES*16*Kp

  const int tid = threadIdx.x, wave = tid >> 5, lane = tid & 31;

  for (int i = tid; i < Np; i += blockDim.x)
    s_bias[i] = (bias && i < N) ? bias[i] : 0.f;
  for (int i = tid; i < Np * Kp; i += blockDim.x) {
    const int n = i / Kp, k = i % Kp;
    s_wt[i] = (_Float16)((n < N && k < K) ? W[(size_t)k * N + n] : 0.f);
  }
  __syncthreads();

  const int mt   = blockIdx.x * GEMM_WAVES + wave;
  const int row0 = mt * 16;
  if (row0 >= M) return;

  if (lane == 0) __builtin_prefetch(X + (size_t)row0 * K, 0, 0); // global_prefetch_b8

  _Float16* xt = s_x + (size_t)wave * 16 * Kp;
  for (int i = lane; i < 16 * Kp; i += 32) {
    const int r = i / Kp, k = i % Kp;
    const int gr = row0 + r;
    float v = 0.f;
    if (gr < M && k < K) v = X[(size_t)gr * K + k];
    xt[i] = (_Float16)v;
  }
  lds_fence();

  for (int nt = 0; nt < Np / 16; ++nt) {
    v8f acc = {};
    for (int kc = 0; kc < Kp / 32; ++kc) {
      v16h a = load_frag(xt + kc * 32, Kp, lane);
      v16h b = load_frag(s_wt + (size_t)nt * 16 * Kp + kc * 32, Kp, lane);
      acc = wmma16(a, b, acc);
    }
    const int col   = nt * 16 + (lane & 15);
    const int rbase = row0 + ((lane >> 4) << 3);
    if (col < N) {
      const float bv = s_bias[nt * 16 + (lane & 15)];
#pragma unroll
      for (int i = 0; i < 8; ++i) {
        const int r = rbase + i;
        if (r < M) {
          float v = acc[i] + bv;
          if (relu && v < 0.f) v = 0.f;
          Y[(size_t)r * N + col] = v;
        }
      }
    }
  }
}

// ---- fused two-layer MLP: Y = act2( relu(X@W1+b1) @ W2 + b2 ) --------------
// Hidden tile lives only in LDS (f16), bounced between the two WMMA passes.
__global__ __launch_bounds__(64) void k_mlp2(
    const float* __restrict__ X,
    const float* __restrict__ W1, const float* __restrict__ b1,
    const float* __restrict__ W2, const float* __restrict__ b2,
    float* __restrict__ Y,
    int M, int K1, int H, int K2, int K1p, int Hp, int K2p, int relu2)
{
  extern __shared__ char smem[];
  float*    sb1 = (float*)smem;                            // Hp
  float*    sb2 = sb1 + Hp;                                // K2p
  _Float16* sw1 = (_Float16*)(sb2 + K2p);                  // Hp*K1p   (W1^T)
  _Float16* sw2 = sw1 + (size_t)Hp * K1p;                  // K2p*Hp   (W2^T)
  _Float16* sx  = sw2 + (size_t)K2p * Hp;                  // MLP_WAVES*16*K1p
  _Float16* shh = sx  + (size_t)MLP_WAVES * 16 * K1p;      // MLP_WAVES*16*Hp

  const int tid = threadIdx.x, wave = tid >> 5, lane = tid & 31;

  for (int i = tid; i < Hp;  i += blockDim.x) sb1[i] = (i < H)  ? b1[i] : 0.f;
  for (int i = tid; i < K2p; i += blockDim.x) sb2[i] = (i < K2) ? b2[i] : 0.f;
  for (int i = tid; i < Hp * K1p; i += blockDim.x) {
    const int h = i / K1p, k = i % K1p;
    sw1[i] = (_Float16)((h < H && k < K1) ? W1[(size_t)k * H + h] : 0.f);
  }
  for (int i = tid; i < K2p * Hp; i += blockDim.x) {
    const int n = i / Hp, h = i % Hp;
    sw2[i] = (_Float16)((n < K2 && h < H) ? W2[(size_t)h * K2 + n] : 0.f);
  }
  __syncthreads();

  const int mt   = blockIdx.x * MLP_WAVES + wave;
  const int row0 = mt * 16;
  if (row0 >= M) return;

  _Float16* xt = sx  + (size_t)wave * 16 * K1p;
  _Float16* ht = shh + (size_t)wave * 16 * Hp;

  for (int i = lane; i < 16 * K1p; i += 32) {
    const int r = i / K1p, k = i % K1p;
    const int gr = row0 + r;
    float v = 0.f;
    if (gr < M && k < K1) v = X[(size_t)gr * K1 + k];
    xt[i] = (_Float16)v;
  }
  lds_fence();

  // layer 1: hidden tile (16 x Hp), relu, into LDS as f16
  for (int htile = 0; htile < Hp / 16; ++htile) {
    v8f acc = {};
    for (int kc = 0; kc < K1p / 32; ++kc) {
      v16h a = load_frag(xt + kc * 32, K1p, lane);
      v16h b = load_frag(sw1 + (size_t)htile * 16 * K1p + kc * 32, K1p, lane);
      acc = wmma16(a, b, acc);
    }
    const int col   = htile * 16 + (lane & 15);
    const int rbase = (lane >> 4) << 3;
    const float bv  = sb1[col];
#pragma unroll
    for (int i = 0; i < 8; ++i) {
      float v = acc[i] + bv;
      if (v < 0.f) v = 0.f;                       // relu (layer 1 always)
      ht[(size_t)(rbase + i) * Hp + col] = (_Float16)v;
    }
  }
  lds_fence();

  // layer 2
  for (int nt = 0; nt < K2p / 16; ++nt) {
    v8f acc = {};
    for (int hc = 0; hc < Hp / 32; ++hc) {
      v16h a = load_frag(ht + hc * 32, Hp, lane);
      v16h b = load_frag(sw2 + (size_t)nt * 16 * Hp + hc * 32, Hp, lane);
      acc = wmma16(a, b, acc);
    }
    const int col   = nt * 16 + (lane & 15);
    const int rbase = row0 + ((lane >> 4) << 3);
    if (col < K2) {
      const float bv = sb2[nt * 16 + (lane & 15)];
#pragma unroll
      for (int i = 0; i < 8; ++i) {
        const int r = rbase + i;
        if (r < M) {
          float v = acc[i] + bv;
          if (relu2 && v < 0.f) v = 0.f;
          Y[(size_t)r * K2 + col] = v;
        }
      }
    }
  }
}

// ---- brute-force kNN-16 (stable, matches top_k(-dist) tie-breaking) --------
// Candidate block is a contiguous raw-f32 copy -> use the CDNA5 async
// global->LDS DMA (no VGPR round trip), drained via s_wait_asynccnt.
__global__ __launch_bounds__(128) void k_knn16(
    const float* __restrict__ qpos, const float* __restrict__ cpos,
    int nq, int ncReal, int ncTot, int* __restrict__ idx)
{
  __shared__ float sc[512 * 3];
  const int b = blockIdx.y;
  const float* gsrc = cpos + (size_t)b * ncReal * 3;
  for (int i = threadIdx.x; i < ncReal * 3; i += blockDim.x) {
    const unsigned ldsA = (unsigned)(uintptr_t)(sc + i);   // low 32b = LDS offset
    async_b32_to_lds(ldsA, (unsigned)(i * 4), gsrc);
  }
  for (int i = ncReal * 3 + threadIdx.x; i < ncTot * 3; i += blockDim.x)
    sc[i] = 0.f;                                           // zero padding
  async_fence();
  __syncthreads();

  const int qi = blockIdx.x * blockDim.x + threadIdx.x;
  if (qi >= nq) return;
  const float qx = qpos[((size_t)b * nq + qi) * 3 + 0];
  const float qy = qpos[((size_t)b * nq + qi) * 3 + 1];
  const float qz = qpos[((size_t)b * nq + qi) * 3 + 2];
  float bd[16]; int bi[16];
#pragma unroll
  for (int j = 0; j < 16; ++j) { bd[j] = 3.4e38f; bi[j] = 0; }
  for (int c = 0; c < ncTot; ++c) {
    const float dx = qx - sc[c * 3 + 0], dy = qy - sc[c * 3 + 1], dz = qz - sc[c * 3 + 2];
    const float d2 = dx * dx + dy * dy + dz * dz;   // same ordering as norm
    if (d2 < bd[15]) {
      int p = 15;
      while (p > 0 && bd[p - 1] > d2) { bd[p] = bd[p - 1]; bi[p] = bi[p - 1]; --p; }
      bd[p] = d2; bi[p] = c;
    }
  }
  for (int j = 0; j < 16; ++j) idx[((size_t)b * nq + qi) * 16 + j] = bi[j];
}

// ---- pair construction: rel_pos, qk_rel, vj --------------------------------
__global__ __launch_bounds__(256) void k_pairs(
    const float* __restrict__ qkv, const float* __restrict__ posq,
    const float* __restrict__ posc, const int* __restrict__ idx,
    float* __restrict__ relpos, float* __restrict__ qk, float* __restrict__ vj,
    int Mq, int J, int d, int nqPerBatch, int ncPerBatch, int mode /*0=idx,1=full*/)
{
  const long p = blockIdx.x * (long)blockDim.x + threadIdx.x;
  if (p >= (long)Mq * J) return;
  const int m = (int)(p / J), j = (int)(p % J);
  const int b = m / nqPerBatch;
  int nbr;
  if (mode == 0) {
    int loc = idx[p];
    if (loc >= ncPerBatch) loc = ncPerBatch - 1;
    nbr = b * ncPerBatch + loc;
  } else {
    nbr = b * ncPerBatch + j;
  }
  for (int c = 0; c < 3; ++c)
    relpos[p * 3 + c] = posq[(size_t)m * 3 + c] - posc[(size_t)nbr * 3 + c];
  const int s3 = 3 * d;
  for (int dd = 0; dd < d; ++dd)
    qk[p * d + dd] = qkv[(size_t)m * s3 + dd] - qkv[(size_t)nbr * s3 + d + dd];
  for (int dd = 0; dd < d; ++dd)
    vj[p * d + dd] = qkv[(size_t)nbr * s3 + 2 * d + dd];
}

// qk += rpe (in place, becomes sim-MLP input); vj += rpe
__global__ __launch_bounds__(256) void k_addpairs(
    float* __restrict__ qk, float* __restrict__ vj,
    const float* __restrict__ rpe, long n)
{
  const long i = blockIdx.x * (long)blockDim.x + threadIdx.x;
  if (i >= n) return;
  const float r = rpe[i];
  qk[i] += r;
  vj[i] += r;
}

// softmax over neighbors (axis j) per channel, then weighted sum with vj
__global__ __launch_bounds__(256) void k_attend(
    const float* __restrict__ sim, const float* __restrict__ vj,
    float* __restrict__ out, int Mq, int J, int d)
{
  const long t = blockIdx.x * (long)blockDim.x + threadIdx.x;
  if (t >= (long)Mq * d) return;
  const int m = (int)(t / d), dd = (int)(t % d);
  float mx = -3.4e38f;
  for (int j = 0; j < J; ++j) {
    const float v = sim[((size_t)m * J + j) * d + dd];
    if (v > mx) mx = v;
  }
  float s = 0.f, o = 0.f;
  for (int j = 0; j < J; ++j) {
    const float e = __expf(sim[((size_t)m * J + j) * d + dd] - mx);
    s += e;
    o += e * vj[((size_t)m * J + j) * d + dd];
  }
  out[t] = o / s;
}

// row gather with JAX-style index clipping
__global__ __launch_bounds__(256) void k_gather(
    const float* __restrict__ src, const int* __restrict__ idx,
    float* __restrict__ out, int outRows, int J, int nqPerBatch,
    int srcPerBatch, int rowLen)
{
  const long t = blockIdx.x * (long)blockDim.x + threadIdx.x;
  if (t >= (long)outRows * rowLen) return;
  const int r = (int)(t / rowLen), c = (int)(t % rowLen);
  const int m = r / J;
  const int b = m / nqPerBatch;
  int loc = idx[r];
  if (loc >= srcPerBatch) loc = srcPerBatch - 1;
  out[t] = src[((size_t)b * srcPerBatch + loc) * rowLen + c];
}

__global__ __launch_bounds__(256) void k_pivot(
    const float* __restrict__ op, const int* __restrict__ perm,
    float* __restrict__ out)
{
  const int t = blockIdx.x * blockDim.x + threadIdx.x;
  if (t >= 8 * 128 * 3) return;
  const int b = t / (128 * 3), rem = t % (128 * 3);
  const int i = rem / 3, c = rem % 3;
  out[t] = op[((size_t)b * 512 + perm[i]) * 3 + c];
}

// (R,48) -> (R,3): reshape(...,3,16).max(-1)
__global__ __launch_bounds__(256) void k_pool(
    const float* __restrict__ in, float* __restrict__ out, int R)
{
  const long t = blockIdx.x * (long)blockDim.x + threadIdx.x;
  if (t >= (long)R * 3) return;
  const int r = (int)(t / 3), dd = (int)(t % 3);
  float mx = -3.4e38f;
  for (int tt = 0; tt < 16; ++tt) {
    const float v = in[(size_t)r * 48 + dd * 16 + tt];
    if (v > mx) mx = v;
  }
  out[t] = mx;
}

__global__ __launch_bounds__(256) void k_concat(
    const float* __restrict__ a, const float* __restrict__ b,
    float* __restrict__ h)
{
  const int t = blockIdx.x * blockDim.x + threadIdx.x;
  if (t >= 4096 * 96) return;
  const int m = t / 96, c = t % 96;
  h[t] = (c < 48) ? a[(size_t)m * 48 + c] : b[(size_t)m * 48 + (c - 48)];
}

// ============================================================================
// Host orchestration
// ============================================================================
static inline int cdiv(int a, int b) { return (a + b - 1) / b; }
static inline int r32(int k) { return ((k + 31) / 32) * 32; }
static inline int r16(int n) { return ((n + 15) / 16) * 16; }

static void gemm1(hipStream_t s, const float* X, const float* W, const float* b,
                  float* Y, int M, int K, int N, int relu) {
  const int Kp = r32(K), Np = r16(N);
  const size_t smem = (size_t)Np * 4 + (size_t)Np * Kp * 2 +
                      (size_t)GEMM_WAVES * 16 * Kp * 2;
  k_gemm1<<<cdiv(M, 16 * GEMM_WAVES), 128, smem, s>>>(X, W, b, Y, M, K, N, Kp, Np, relu);
}

static void mlp2(hipStream_t s, const float* X, const float* W1, const float* b1,
                 const float* W2, const float* b2, float* Y,
                 int M, int K1, int H, int K2, int relu2) {
  const int K1p = r32(K1), Hp = r32(H), K2p = r16(K2);
  const size_t smem = (size_t)(Hp + K2p) * 4 + (size_t)Hp * K1p * 2 +
                      (size_t)K2p * Hp * 2 + (size_t)MLP_WAVES * 16 * K1p * 2 +
                      (size_t)MLP_WAVES * 16 * Hp * 2;
  k_mlp2<<<cdiv(M, 16 * MLP_WAVES), 64, smem, s>>>(X, W1, b1, W2, b2, Y,
                                                   M, K1, H, K2, K1p, Hp, K2p, relu2);
}

extern "C" void kernel_launch(void* const* d_in, const int* in_sizes, int n_in,
                              void* d_out, int out_size, void* d_ws, size_t ws_size,
                              hipStream_t stream) {
  (void)in_sizes; (void)n_in; (void)out_size; (void)ws_size;

  const float* OP   = (const float*)d_in[0];   // (8,512,3)
  const float* DATA = (const float*)d_in[1];   // (8,512,16,3)
  const int*   PERM = (const int*)d_in[2];     // (128,)
  // params flattened in setup_inputs() insertion order: per layer i (a1..a5):
  // wqkv, pw1, pb1, pw2, pb2, aw1, ab1, aw2, ab2  — then mlp/out params.
  auto Lp = [&](int layer, int j) { return (const float*)d_in[3 + layer * 9 + j]; };
  const float* MLP_W1 = (const float*)d_in[48];
  const float* MLP_B1 = (const float*)d_in[49];
  const float* MLP_W2 = (const float*)d_in[50];
  const float* MLP_B2 = (const float*)d_in[51];
  const float* OUT_W1 = (const float*)d_in[52];
  const float* OUT_B1 = (const float*)d_in[53];
  const float* OUT_W2 = (const float*)d_in[54];
  const float* OUT_B2 = (const float*)d_in[55];
  const float* OUT_W3 = (const float*)d_in[56];
  const float* OUT_B3 = (const float*)d_in[57];

  float* ws = (float*)d_ws;
  size_t off = 0;
  auto alloc = [&](size_t n) { float* p = ws + off; off += n; return p; };
  float* sa2   = alloc(4096u * 48);
  float* sa3   = alloc(16384u * 3);
  float* sa4   = alloc(1024u * 48);
  float* pivot = alloc(1024u * 3);
  float* t1    = alloc(4096u * 48);
  float* t2    = alloc(4096u * 48);
  float* pool  = alloc(65536u * 3);
  float* nbpts = alloc(16384u * 3);
  float* uppts = alloc(65536u * 3);
  float* qkvb  = alloc(65536u * 9);            // 589,824 (also fits 4096x144)
  float* h96   = alloc(4096u * 96);
  int*   idxa2 = (int*)alloc(4096u * 16);
  int*   idxd  = (int*)alloc(1024u * 16);
  int*   idxu  = (int*)alloc(4096u * 16);
  float* relpos = alloc(1048576u * 3);         // pair arena, reused per layer
  float* qk     = alloc(131072u * 48);
  float* vj     = alloc(131072u * 48);
  float* rpe    = alloc(131072u * 48);
  float* gath   = relpos;                      // gathers happen while arena is idle

  auto pt_tail = [&](int layer, int Mq, int J, int d, long P, float* out) {
    // rpe MLP (3 -> 64 -> d), fused
    mlp2(stream, relpos, Lp(layer, 1), Lp(layer, 2), Lp(layer, 3), Lp(layer, 4),
         rpe, (int)P, 3, 64, d, 0);
    k_addpairs<<<cdiv((int)(P * d), 256), 256, 0, stream>>>(qk, vj, rpe, P * d);
    // sim MLP (d -> 4d -> d), fused; output into rpe buffer (free now)
    mlp2(stream, qk, Lp(layer, 5), Lp(layer, 6), Lp(layer, 7), Lp(layer, 8),
         rpe, (int)P, d, 4 * d, d, 0);
    k_attend<<<cdiv(Mq * d, 256), 256, 0, stream>>>(rpe, vj, out, Mq, J, d);
  };

  // ---------------- a1: grouped (4096 groups of 16), dim 3 ----------------
  gemm1(stream, DATA, Lp(0, 0), nullptr, qkvb, 65536, 3, 9, 0);
  k_pairs<<<cdiv(1048576, 256), 256, 0, stream>>>(qkvb, DATA, DATA, nullptr,
      relpos, qk, vj, 65536, 16, 3, 16, 16, 1);
  pt_tail(0, 65536, 16, 3, 1048576, t1);        // t1 == sa1 as (4096,48)
  gemm1(stream, t1, MLP_W1, MLP_B1, t2, 4096, 48, 48, 1);
  gemm1(stream, t2, MLP_W2, MLP_B2, t1, 4096, 48, 48, 1);

  // ---------------- a2: kNN-16 over 512 pts/batch, dim 48 ----------------
  gemm1(stream, t1, Lp(1, 0), nullptr, qkvb, 4096, 48, 144, 0);
  k_knn16<<<dim3(cdiv(512, 128), 8), 128, 0, stream>>>(OP, OP, 512, 512, 512, idxa2);
  k_pairs<<<cdiv(65536, 256), 256, 0, stream>>>(qkvb, OP, OP, idxa2,
      relpos, qk, vj, 4096, 16, 48, 512, 512, 0);
  pt_tail(1, 4096, 16, 48, 65536, t2);
  gemm1(stream, t2, MLP_W1, MLP_B1, t1, 4096, 48, 48, 1);
  gemm1(stream, t1, MLP_W2, MLP_B2, sa2, 4096, 48, 48, 1);

  // ---------------- downsample: pivot, idx_d, gather, pool ----------------
  k_pivot<<<cdiv(3072, 256), 256, 0, stream>>>(OP, PERM, pivot);
  k_knn16<<<dim3(1, 8), 128, 0, stream>>>(pivot, OP, 128, 512, 512, idxd);
  k_gather<<<cdiv(16384 * 48, 256), 256, 0, stream>>>(sa2, idxd, gath, 16384, 16, 128, 512, 48);
  k_pool<<<cdiv(16384 * 3, 256), 256, 0, stream>>>(gath, pool, 16384);
  k_gather<<<cdiv(16384 * 3, 256), 256, 0, stream>>>(OP, idxd, nbpts, 16384, 16, 128, 512, 3);

  // ---------------- a3: grouped (1024 groups of 16), dim 3 ----------------
  gemm1(stream, pool, Lp(2, 0), nullptr, qkvb, 16384, 3, 9, 0);
  k_pairs<<<cdiv(262144, 256), 256, 0, stream>>>(qkvb, nbpts, nbpts, nullptr,
      relpos, qk, vj, 16384, 16, 3, 16, 16, 1);
  pt_tail(2, 16384, 16, 3, 262144, sa3);        // sa3 == (1024,48)

  // ---------------- a4: full attention over 128, dim 48 ----------------
  gemm1(stream, sa3, Lp(3, 0), nullptr, qkvb, 1024, 48, 144, 0);
  k_pairs<<<cdiv(131072, 256), 256, 0, stream>>>(qkvb, pivot, pivot, nullptr,
      relpos, qk, vj, 1024, 128, 48, 128, 128, 1);
  pt_tail(3, 1024, 128, 48, 131072, sa4);

  // ---------------- upsample: idx_u (padded pivots), gather, pool ----------
  k_knn16<<<dim3(cdiv(512, 128), 8), 128, 0, stream>>>(OP, pivot, 512, 128, 512, idxu);
  k_gather<<<cdiv(65536 * 48, 256), 256, 0, stream>>>(sa4, idxu, gath, 65536, 16, 512, 128, 48);
  k_pool<<<cdiv(65536 * 3, 256), 256, 0, stream>>>(gath, pool, 65536);
  k_gather<<<cdiv(65536 * 3, 256), 256, 0, stream>>>(pivot, idxu, uppts, 65536, 16, 512, 128, 3);

  // ---------------- a5: grouped (4096 groups of 16), dim 3 ----------------
  gemm1(stream, pool, Lp(4, 0), nullptr, qkvb, 65536, 3, 9, 0);
  k_pairs<<<cdiv(1048576, 256), 256, 0, stream>>>(qkvb, uppts, uppts, nullptr,
      relpos, qk, vj, 65536, 16, 3, 16, 16, 1);
  pt_tail(4, 65536, 16, 3, 1048576, t1);        // t1 == sa5 as (4096,48)

  // ---------------- head ----------------
  k_concat<<<cdiv(4096 * 96, 256), 256, 0, stream>>>(t1, sa2, h96);
  gemm1(stream, h96, OUT_W1, OUT_B1, t2, 4096, 96, 48, 1);
  gemm1(stream, t2, OUT_W2, OUT_B2, t1, 4096, 48, 48, 1);
  gemm1(stream, t1, OUT_W3, OUT_B3, (float*)d_out, 4096, 48, 1, 0);
}
